// MoDSurpriseRouter_37778532336199
// MI455X (gfx1250) — matrix operand order repaired and checked by reading, not compiled, not run on hardware
//
#include <hip/hip_runtime.h>

// ---------------------------------------------------------------------------
// MoD surprise router for MI455X (gfx1250, wave32, WMMA + TDM).
// All GEMM-shaped math runs through v_wmma_f32_16x16x32_bf16 (fp32 accum).
// k_gemm stages its weight panel into LDS with tensor_load_to_lds (TDM),
// double-buffered, ordered by s_wait_tensorcnt + workgroup barrier.
// ---------------------------------------------------------------------------

#define BATCH  256
#define NPATCH 256
#define DIM    384
#define DH     256
#define KTOP   76
#define KPAD   80
#define BN     (BATCH * NPATCH)       // 65536
#define BND    ((size_t)BN * DIM)     // 25165824
#define MH     (BATCH * KPAD)         // 20480 heavy tokens (padded)

typedef __attribute__((ext_vector_type(8)))  float  v8f;
typedef __attribute__((ext_vector_type(16))) __bf16 v16bf;
typedef __attribute__((ext_vector_type(4)))  unsigned int u32x4;
typedef __attribute__((ext_vector_type(4)))  int i32x4;
typedef __attribute__((ext_vector_type(8)))  int i32x8;

#if defined(__has_builtin)
#if __has_builtin(__builtin_amdgcn_tensor_load_to_lds) && \
    __has_builtin(__builtin_amdgcn_s_wait_tensorcnt)
#define USE_TDM 1
#endif
#endif
#ifndef USE_TDM
#define USE_TDM 0
#endif

union FragU { v16bf v; float4 q[2]; };

__device__ __forceinline__ v8f wmma_bf16(v16bf a, v16bf b, v8f c) {
  return __builtin_amdgcn_wmma_f32_16x16x32_bf16(false, a, false, b, (short)0, c,
                                                 false, false);
}
// A-fragment: 16 contiguous bf16 at p (K=base16..base16+15 for row m)
__device__ __forceinline__ v16bf ldA(const __bf16* p) {
  FragU f; f.q[0] = *(const float4*)p; f.q[1] = *(const float4*)(p + 8); return f.v;
}
// B-fragment: 16 contiguous bf16 at p (K=base16..base16+15 for column n)
__device__ __forceinline__ v16bf ldB(const __bf16* p) {
  FragU f; f.q[0] = *(const float4*)p; f.q[1] = *(const float4*)(p + 8); return f.v;
}
__device__ __forceinline__ v16bf zfrag() {
  FragU f; f.q[0] = make_float4(0.f, 0.f, 0.f, 0.f); f.q[1] = f.q[0]; return f.v;
}
__device__ __forceinline__ v8f zacc() {
  v8f z = {0.f, 0.f, 0.f, 0.f, 0.f, 0.f, 0.f, 0.f}; return z;
}

#if USE_TDM
// 1-D TDM copy: nbytes (even) from global gsrc -> LDS byte offset lds_addr.
// D# per CDNA5 ISA: group0 = {count=1, lds_addr, gaddr, type=2},
// group1 = {data_size=2B, tensor_dim0=tile_dim0=nelem, tensor_dim1=tile_dim1=1}.
__device__ __forceinline__ void tdm_load_1d(unsigned lds_addr, const void* gsrc,
                                            unsigned nbytes) {
  unsigned long long ga = (unsigned long long)(uintptr_t)gsrc;
  unsigned ne = nbytes >> 1;  // elements of 2 bytes
  u32x4 g0;
  g0[0] = 1u;                                   // count=1, user descriptor
  g0[1] = lds_addr;                             // LDS byte address
  g0[2] = (unsigned)(ga & 0xFFFFFFFFu);         // global_addr[31:0]
  g0[3] = (unsigned)((ga >> 32) & 0x1FFFFFFu) | (2u << 30);  // [56:32] | type=2
  i32x8 g1;
  g1[0] = (int)(1u << 16);                      // data_size=1 (2 bytes/elem)
  g1[1] = (int)((ne & 0xFFFFu) << 16);          // tensor_dim0[15:0]
  g1[2] = (int)(((ne >> 16) & 0xFFFFu) | (1u << 16));  // dim0[31:16] | dim1=1
  g1[3] = (int)((ne & 0xFFFFu) << 16);          // dim1[31:16]=0 | tile_dim0
  g1[4] = 1;                                    // tile_dim1=1, tile_dim2=0
  g1[5] = (int)ne;                              // tensor_dim0_stride[31:0]
  g1[6] = 0;
  g1[7] = 0;
  i32x4 z4 = {0, 0, 0, 0};
#if __clang_major__ >= 23
  i32x8 z8 = {0, 0, 0, 0, 0, 0, 0, 0};
  __builtin_amdgcn_tensor_load_to_lds(g0, g1, z4, z4, z8, 0);
#else
  __builtin_amdgcn_tensor_load_to_lds(g0, g1, z4, z4, 0);
#endif
}
#endif

// ------------------------------- pack kernels ------------------------------

__global__ __launch_bounds__(256) void k_cast_bf16(const float* __restrict__ s,
                                                   __bf16* __restrict__ d, int n) {
  int i = blockIdx.x * 256 + threadIdx.x;
  if (i < n) d[i] = (__bf16)s[i];
}

// W[OUT][IN] -> dst[IN/32][OUT][32]  (B-fragment friendly, K-major steps)
__global__ __launch_bounds__(256) void k_pack_linw(const float* __restrict__ w,
                                                   __bf16* __restrict__ dst,
                                                   int OUT, int IN) {
  int i = blockIdx.x * 256 + threadIdx.x;
  if (i >= OUT * IN) return;
  int n = i % OUT, kidx = i / OUT;
  int kk = kidx >> 5, kin = kidx & 31;
  dst[((size_t)kk * OUT + n) * 32 + kin] = (__bf16)w[(size_t)n * IN + kidx];
}

// conv_w[O=384][I=384][3][3] -> dst[108][384][32] with k = tap*384 + cin
__global__ __launch_bounds__(256) void k_pack_convw(const float* __restrict__ w,
                                                    __bf16* __restrict__ dst) {
  int i = blockIdx.x * 256 + threadIdx.x;
  if (i >= 3456 * 384) return;
  int n = i % 384, k = i / 384;
  int tap = k / 384, cin = k % 384;
  int ky = tap / 3, kx = tap % 3;
  int kk = k >> 5, kin = k & 31;
  dst[((size_t)kk * 384 + n) * 32 + kin] =
      (__bf16)w[(((size_t)n * 384 + cin) * 3 + ky) * 3 + kx];
}

// ---------------------- fused conv3x3 + bias + GN (+SiLU) ------------------
// grid (8 groups, BATCH); 256 thr = 8 waves; wave owns 2 M-tiles x 3 N-tiles.
// K loop = tap(9) x cstep(12): tap decode + border predicates hoisted.
// Last layer: psq != nullptr -> per-patch partial sum of (pred - tgt)^2.
__global__ __launch_bounds__(256) void k_conv_gn(
    const __bf16* __restrict__ actIn, const __bf16* __restrict__ wpack,
    const float* __restrict__ conv_b, const float* __restrict__ gn_g,
    const float* __restrict__ gn_b, __bf16* __restrict__ actOut,
    float* __restrict__ psq, const float* __restrict__ tgt, int do_silu) {
  int g = blockIdx.x, b = blockIdx.y;
  int tid = threadIdx.x, wave = tid >> 5, lane = tid & 31;
  int lhalf = lane >> 4, l16 = lane & 15;
  int mt0 = 2 * wave, mt1 = 2 * wave + 1;

  v8f acc[6];
#pragma unroll
  for (int t = 0; t < 6; t++) acc[t] = zacc();

  // lane-invariant part of the weight pointer: column (g*48 + l16), K half
  const __bf16* wbase = wpack + ((size_t)(g * 48 + l16)) * 32 + lhalf * 16;
  size_t kkoff = 0;  // advances 384*32 elements per K-step

  for (int tap = 0; tap < 9; ++tap) {
    int dy = tap / 3 - 1, dx = tap % 3 - 1;  // 9x only, cheap
    const __bf16* ap[2];
    bool av[2];
#pragma unroll
    for (int q = 0; q < 2; q++) {
      int m = (q ? mt1 : mt0) * 16 + l16;
      int ny = (m >> 4) + dy, nx = (m & 15) + dx;
      av[q] = (ny >= 0) && (ny < 16) && (nx >= 0) && (nx < 16);
      ap[q] = actIn + ((size_t)(b * NPATCH + ny * 16 + nx)) * DIM + lhalf * 16;
    }
    // prefetch next tap's weight panel (global_prefetch_b8)
    __builtin_prefetch((const void*)(wbase + kkoff + (size_t)12 * 12288), 0, 1);
    for (int cs = 0; cs < 12; ++cs) {
      v16bf afr0 = av[0] ? ldA(ap[0] + cs * 32) : zfrag();
      v16bf afr1 = av[1] ? ldA(ap[1] + cs * 32) : zfrag();
      const __bf16* wp = wbase + kkoff;
#pragma unroll
      for (int nt = 0; nt < 3; nt++) {
        v16bf bfr = ldB(wp + nt * 512);
        acc[0 * 3 + nt] = wmma_bf16(afr0, bfr, acc[0 * 3 + nt]);
        acc[1 * 3 + nt] = wmma_bf16(afr1, bfr, acc[1 * 3 + nt]);
      }
      kkoff += 12288;  // 384*32 elements per K-step
    }
  }

  // bias + per-lane stats
  float ls = 0.f, lsq = 0.f;
#pragma unroll
  for (int q = 0; q < 2; q++)
#pragma unroll
    for (int nt = 0; nt < 3; nt++) {
      int n = g * 48 + nt * 16 + l16;
      float bb = conv_b[n];
#pragma unroll
      for (int r = 0; r < 8; r++) {
        float v = acc[q * 3 + nt][r] + bb;
        acc[q * 3 + nt][r] = v;
        ls += v; lsq += v * v;
      }
    }
#pragma unroll
  for (int o = 16; o > 0; o >>= 1) {
    ls += __shfl_xor(ls, o, 32);
    lsq += __shfl_xor(lsq, o, 32);
  }
  __shared__ float red[16];
  __shared__ float stats[2];
  __shared__ float predl[NPATCH * 48];  // used only on last layer
  if (lane == 0) { red[wave] = ls; red[8 + wave] = lsq; }
  __syncthreads();
  if (tid == 0) {
    float S = 0.f, Q = 0.f;
    for (int w = 0; w < 8; w++) { S += red[w]; Q += red[8 + w]; }
    float mu = S * (1.f / 12288.f);
    float var = Q * (1.f / 12288.f) - mu * mu;
    stats[0] = mu; stats[1] = rsqrtf(var + 1e-5f);
  }
  __syncthreads();
  float mu = stats[0], rs = stats[1];

#pragma unroll
  for (int q = 0; q < 2; q++)
#pragma unroll
    for (int nt = 0; nt < 3; nt++) {
      int n = g * 48 + nt * 16 + l16;
      float gg = gn_g[n], gb = gn_b[n];
#pragma unroll
      for (int r = 0; r < 8; r++) {
        int m = (q ? mt1 : mt0) * 16 + r + lhalf * 8;
        float v = (acc[q * 3 + nt][r] - mu) * rs * gg + gb;
        if (do_silu) v = v / (1.f + expf(-v));
        if (actOut) actOut[((size_t)(b * NPATCH + m)) * DIM + n] = (__bf16)v;
        if (psq) predl[m * 48 + (n - g * 48)] = v;
      }
    }
  if (psq) {
    __syncthreads();
    int p = tid;  // 256 threads, 256 patches
    const float* trow = tgt + ((size_t)(b * NPATCH + p)) * DIM + g * 48;
    float s = 0.f;
#pragma unroll 8
    for (int c = 0; c < 48; c++) {
      float d = predl[p * 48 + c] - trow[c];
      s += d * d;
    }
    psq[(size_t)g * BN + b * NPATCH + p] = s;
  }
}

// ------------------------------ generic GEMM -------------------------------
enum { EP_BF16 = 0, EP_SILU = 1, EP_GELU = 2, EP_F32 = 3, EP_F32RES = 4 };

// out[M,Nt] = act( A[M,Kd] @ W + bias )  (W packed [Kd/32][Nt][32])
// grid (M/128, Nt/64), 256 thr: wave owns 16 rows x 64 cols (4 tiles).
// B panel (contiguous 64x32 bf16 = 4KB per K-step) is staged to LDS by the
// Tensor Data Mover, double-buffered; s_wait_tensorcnt + barrier order it.
__global__ __launch_bounds__(256) void k_gemm(
    const __bf16* __restrict__ A, const __bf16* __restrict__ Bp,
    const float* __restrict__ bias, float* __restrict__ outF,
    __bf16* __restrict__ outB, const float* __restrict__ res,
    int M, int Kd, int Nt, int mode) {
  int wave = threadIdx.x >> 5, lane = threadIdx.x & 31;
  int lhalf = lane >> 4, l16 = lane & 15;
  int m0 = blockIdx.x * 128 + wave * 16;
  int nb = blockIdx.y * 64;
  v8f acc[4];
#pragma unroll
  for (int t = 0; t < 4; t++) acc[t] = zacc();
  int steps = Kd >> 5;
  const __bf16* arow = A + (size_t)(m0 + l16) * Kd + lhalf * 16;

#if USE_TDM
  __shared__ __bf16 bpanel[2][64 * 32];  // 2 x 4KB double buffer
  unsigned lds0 = (unsigned)(uintptr_t)&bpanel[0][0];
  unsigned lds1 = (unsigned)(uintptr_t)&bpanel[1][0];
  bool issuer = (threadIdx.x < 32);  // wave 0 drives the TDM
  if (issuer) tdm_load_1d(lds0, Bp + (size_t)nb * 32, 64 * 32 * 2);
  for (int kk = 0; kk < steps; ++kk) {
    __builtin_amdgcn_s_wait_tensorcnt(0);  // oldest TDM op done (no-op elsewhere)
    __syncthreads();                       // publish panel kk to all waves
    if (issuer && (kk + 1) < steps)
      tdm_load_1d((kk & 1) ? lds0 : lds1,
                  Bp + ((size_t)(kk + 1) * Nt + nb) * 32, 64 * 32 * 2);
    v16bf a = ldA(arow + kk * 32);
    const __bf16* pan = &bpanel[kk & 1][0];
#pragma unroll
    for (int nt = 0; nt < 4; nt++) {
      v16bf bfr = ldB(pan + (nt * 16 + l16) * 32 + lhalf * 16);
      acc[nt] = wmma_bf16(a, bfr, acc[nt]);
    }
  }
#else
  for (int kk = 0; kk < steps; ++kk) {
    v16bf a = ldA(arow + kk * 32);
#pragma unroll
    for (int nt = 0; nt < 4; nt++) {
      v16bf bfr = ldB(Bp + ((size_t)kk * Nt + nb + nt * 16 + l16) * 32 + lhalf * 16);
      acc[nt] = wmma_bf16(a, bfr, acc[nt]);
    }
  }
#endif

#pragma unroll
  for (int nt = 0; nt < 4; nt++) {
    int n = nb + nt * 16 + l16;
    float bb = bias[n];
#pragma unroll
    for (int r = 0; r < 8; r++) {
      int m = m0 + r + lhalf * 8;
      float v = acc[nt][r] + bb;
      if (mode == EP_SILU) v = v / (1.f + expf(-v));
      else if (mode == EP_GELU) v = 0.5f * v * (1.f + erff(v * 0.70710678118f));
      size_t o = (size_t)m * Nt + n;
      if (mode == EP_F32) outF[o] = v;
      else if (mode == EP_F32RES) outF[o] = v + res[o];
      else outB[o] = (__bf16)v;
    }
  }
}

// ------------------------------- layernorm ---------------------------------
// one wave per token; D in {256, 384}; output bf16 (GEMM A operand)
__global__ __launch_bounds__(256) void k_layernorm(
    const float* __restrict__ X, const float* __restrict__ g,
    const float* __restrict__ b, __bf16* __restrict__ Y, int M, int D) {
  int wave = threadIdx.x >> 5, lane = threadIdx.x & 31;
  int t = blockIdx.x * 8 + wave;
  if (t >= M) return;
  const float* x = X + (size_t)t * D;
  int nv = D >> 5;
  float vals[12];
  float s = 0.f;
  for (int i = 0; i < nv; i++) { vals[i] = x[lane + i * 32]; s += vals[i]; }
#pragma unroll
  for (int o = 16; o > 0; o >>= 1) s += __shfl_xor(s, o, 32);
  float muv = s / (float)D;
  float q = 0.f;
  for (int i = 0; i < nv; i++) { float d = vals[i] - muv; q += d * d; }
#pragma unroll
  for (int o = 16; o > 0; o >>= 1) q += __shfl_xor(q, o, 32);
  float rs = rsqrtf(q / (float)D + 1e-5f);
  for (int i = 0; i < nv; i++) {
    int c = lane + i * 32;
    Y[(size_t)t * D + c] = (__bf16)((vals[i] - muv) * rs * g[c] + b[c]);
  }
}

// ----------------------- surprise + deterministic loss ---------------------
__global__ __launch_bounds__(256) void k_surprise(const float* __restrict__ psq,
                                                  float* __restrict__ surp,
                                                  float* __restrict__ lpart) {
  int i = blockIdx.x * 256 + threadIdx.x;  // grid 256 -> covers BN
  float s = 0.f;
#pragma unroll
  for (int g = 0; g < 8; g++) s += psq[(size_t)g * BN + i];
  s *= (1.f / (float)DIM);
  surp[i] = s;
  __shared__ float red[256];
  red[threadIdx.x] = s;
  __syncthreads();
  for (int o = 128; o > 0; o >>= 1) {
    if (threadIdx.x < o) red[threadIdx.x] += red[threadIdx.x + o];
    __syncthreads();
  }
  if (threadIdx.x == 0) lpart[blockIdx.x] = red[0];
}

__global__ __launch_bounds__(256) void k_loss(const float* __restrict__ lpart,
                                              float* __restrict__ loss) {
  __shared__ float red[256];
  red[threadIdx.x] = lpart[threadIdx.x];
  __syncthreads();
  for (int o = 128; o > 0; o >>= 1) {
    if (threadIdx.x < o) red[threadIdx.x] += red[threadIdx.x + o];
    __syncthreads();
  }
  if (threadIdx.x == 0) loss[0] = red[0] * (1.f / (float)BN);
}

// --------------------- per-row top-K via bitonic sort ----------------------
// key: value desc, index asc (matches lax.top_k tie behavior)
__global__ __launch_bounds__(256) void k_topk(const float* __restrict__ surp,
                                              int* __restrict__ topk,
                                              float* __restrict__ mask) {
  int b = blockIdx.x, t = threadIdx.x;
  __shared__ float v[256];
  __shared__ int ix[256];
  __shared__ int flag[256];
  v[t] = surp[b * NPATCH + t];
  ix[t] = t;
  flag[t] = 0;
  __syncthreads();
  for (int k = 2; k <= 256; k <<= 1) {
    for (int j = k >> 1; j > 0; j >>= 1) {
      int p = t ^ j;
      if (p > t) {
        bool up = ((t & k) == 0);  // "best-first" blocks
        float va = v[t], vb = v[p];
        int ia = ix[t], ib = ix[p];
        bool before = (va > vb) || (va == vb && ia < ib);
        if (up ? !before : before) { v[t] = vb; v[p] = va; ix[t] = ib; ix[p] = ia; }
      }
      __syncthreads();
    }
  }
  if (t < KTOP) flag[ix[t]] = 1;
  if (t < KPAD) topk[b * KPAD + t] = (t < KTOP) ? ix[t] : -1;
  __syncthreads();
  mask[b * NPATCH + t] = (float)flag[t];
}

// ------------------------------ gather/scatter -----------------------------
__global__ __launch_bounds__(128) void k_gather(const float* __restrict__ feat,
                                                const int* __restrict__ topk,
                                                __bf16* __restrict__ tok) {
  int r = blockIdx.x;  // 0..MH-1
  int b = r / KPAD;
  int idx = topk[r];
  __bf16* dst = tok + (size_t)r * DIM;
  if (idx >= 0) {
    const float* src = feat + ((size_t)b * NPATCH + idx) * DIM;
    for (int c = threadIdx.x; c < DIM; c += 128) dst[c] = (__bf16)src[c];
  } else {
    for (int c = threadIdx.x; c < DIM; c += 128) dst[c] = (__bf16)0.f;
  }
}

__global__ __launch_bounds__(128) void k_scatter(const float* __restrict__ hout,
                                                 const int* __restrict__ topk,
                                                 float* __restrict__ routed) {
  int b = blockIdx.x, k = blockIdx.y;
  int idx = topk[b * KPAD + k];
  if (idx < 0) return;
  const float* src = hout + ((size_t)(b * KPAD + k)) * DIM;
  float* dst = routed + ((size_t)b * NPATCH + idx) * DIM;
  for (int c = threadIdx.x; c < DIM; c += 128) dst[c] = src[c];
}

// --------------------------- fused MHA (per b,head) ------------------------
// qkv bf16 [MH][768]; out bf16 [MH][256]. Padded keys 76..79 masked.
__global__ __launch_bounds__(256) void k_attn(const __bf16* __restrict__ qkv,
                                              __bf16* __restrict__ outb) {
  int h = blockIdx.x, b = blockIdx.y;
  int tid = threadIdx.x, wave = tid >> 5, lane = tid & 31;
  int lhalf = lane >> 4, l16 = lane & 15;
  __shared__ float S[80 * 80];
  __shared__ __bf16 P[80 * 96];
  __shared__ __bf16 VT[64 * 96];
  size_t base = (size_t)b * KPAD * 768;
  // stage V^T (token-contiguous per output dim), zero-pad tokens 80..95
  for (int i = tid; i < 64 * 96; i += 256) {
    int d = i / 96, t = i % 96;
    VT[i] = (t < KPAD) ? qkv[base + (size_t)t * 768 + 512 + h * 64 + d] : (__bf16)0.f;
  }
  // scores = Q @ K^T / 8, 5x5 tiles, K=64 (2 steps)
  for (int t = wave; t < 25; t += 8) {
    int mt = t / 5, nt = t % 5;
    v8f acc = zacc();
#pragma unroll
    for (int kk = 0; kk < 2; kk++) {
      v16bf a = ldA(qkv + base + (size_t)(mt * 16 + l16) * 768 + h * 64 + kk * 32 +
                    lhalf * 16);
      v16bf bfr = ldB(qkv + base + (size_t)(nt * 16 + l16) * 768 + 256 + h * 64 +
                      kk * 32 + lhalf * 16);
      acc = wmma_bf16(a, bfr, acc);
    }
#pragma unroll
    for (int r = 0; r < 8; r++) {
      int row = mt * 16 + r + lhalf * 8;
      int col = nt * 16 + l16;
      float vv = acc[r] * 0.125f;
      if (col >= KTOP) vv = -1e30f;
      S[row * 80 + col] = vv;
    }
  }
  __syncthreads();
  // softmax rows (one thread per row), write bf16 probs zero-padded to 96
  if (tid < 80) {
    float mx = -1e30f;
    for (int c = 0; c < 80; c++) mx = fmaxf(mx, S[tid * 80 + c]);
    float sum = 0.f;
    for (int c = 0; c < 80; c++) {
      float e = expf(S[tid * 80 + c] - mx);
      sum += e;
      S[tid * 80 + c] = e;
    }
    float inv = 1.f / sum;
    for (int c = 0; c < 96; c++)
      P[tid * 96 + c] = (c < 80) ? (__bf16)(S[tid * 80 + c] * inv) : (__bf16)0.f;
  }
  __syncthreads();
  // O = P @ V : 5x4 tiles, K=96 (3 steps), A from LDS P, B from LDS VT
  for (int t = wave; t < 20; t += 8) {
    int mt = t / 4, nt = t % 4;
    v8f acc = zacc();
#pragma unroll
    for (int kk = 0; kk < 3; kk++) {
      v16bf a = ldA((const __bf16*)&P[(mt * 16 + l16) * 96 + kk * 32 + lhalf * 16]);
      v16bf bfr = ldB((const __bf16*)&VT[(nt * 16 + l16) * 96 + kk * 32 + lhalf * 16]);
      acc = wmma_bf16(a, bfr, acc);
    }
#pragma unroll
    for (int r = 0; r < 8; r++) {
      int row = mt * 16 + r + lhalf * 8;
      int d = nt * 16 + l16;
      outb[(size_t)(b * KPAD + row) * DH + h * 64 + d] = (__bf16)acc[r];
    }
  }
}

// =============================== launch ====================================

extern "C" void kernel_launch(void* const* d_in, const int* in_sizes, int n_in,
                              void* d_out, int out_size, void* d_ws, size_t ws_size,
                              hipStream_t stream) {
  (void)in_sizes; (void)n_in; (void)out_size; (void)ws_size;
  // --------- inputs (jax pytree = sorted dict keys) ---------
  const float* feat_cur = (const float*)d_in[0];
  const float* feat_next = (const float*)d_in[1];
  int ii = 2;
  const float *fwm_cb[3], *fwm_cw[3], *fwm_gb[3], *fwm_gg[3];
  for (int l = 0; l < 3; l++) {  // sorted: conv_b, conv_w, gn_b, gn_g
    fwm_cb[l] = (const float*)d_in[ii++]; fwm_cw[l] = (const float*)d_in[ii++];
    fwm_gb[l] = (const float*)d_in[ii++]; fwm_gg[l] = (const float*)d_in[ii++];
  }
  // heavy.layers[l]: ff1(b,w), ff2(b,w), in_proj_b, in_proj_w, ln1_b, ln1_g,
  //                  ln2_b, ln2_g, out_proj(b,w)
  const float *ff1_b[2], *ff1_w[2], *ff2_b[2], *ff2_w[2], *inp_b[2], *inp_w[2];
  const float *l1b[2], *l1g[2], *l2b[2], *l2g[2], *op_b[2], *op_w[2];
  for (int l = 0; l < 2; l++) {
    ff1_b[l] = (const float*)d_in[ii++]; ff1_w[l] = (const float*)d_in[ii++];
    ff2_b[l] = (const float*)d_in[ii++]; ff2_w[l] = (const float*)d_in[ii++];
    inp_b[l] = (const float*)d_in[ii++]; inp_w[l] = (const float*)d_in[ii++];
    l1b[l] = (const float*)d_in[ii++];  l1g[l] = (const float*)d_in[ii++];
    l2b[l] = (const float*)d_in[ii++];  l2g[l] = (const float*)d_in[ii++];
    op_b[l] = (const float*)d_in[ii++]; op_w[l] = (const float*)d_in[ii++];
  }
  const float* pin_b = (const float*)d_in[ii++];
  const float* pin_w = (const float*)d_in[ii++];
  const float* pout_b = (const float*)d_in[ii++];
  const float* pout_w = (const float*)d_in[ii++];
  const float* lin1_b = (const float*)d_in[ii++];
  const float* lin1_w = (const float*)d_in[ii++];
  const float* lin2_b = (const float*)d_in[ii++];
  const float* lin2_w = (const float*)d_in[ii++];
  const float* lln_b = (const float*)d_in[ii++];
  const float* lln_g = (const float*)d_in[ii++];

  // --------- outputs ---------
  float* routed = (float*)d_out;
  float* surp = routed + BND;
  float* mask = surp + BN;
  float* loss = mask + BN;

  // --------- workspace layout ---------
  char* base = (char*)d_ws;
  size_t off = 0;
  auto alloc = [&](size_t bytes) -> void* {
    void* p = base + off;
    off = (off + bytes + 255) & ~(size_t)255;
    return p;
  };
  __bf16* actA = (__bf16*)alloc(BND * 2);
  __bf16* actB = (__bf16*)alloc(BND * 2);
  __bf16* convW[3];
  for (int l = 0; l < 3; l++) convW[l] = (__bf16*)alloc((size_t)3456 * 384 * 2);
  float* psq = (float*)alloc((size_t)8 * BN * 4);
  float* lpart = (float*)alloc(256 * 4);
  int* topk = (int*)alloc((size_t)BATCH * KPAD * 4);
  __bf16* tok = (__bf16*)alloc((size_t)MH * DIM * 2);
  float* hbuf = (float*)alloc((size_t)MH * DH * 4);
  __bf16* lnbuf = (__bf16*)alloc((size_t)MH * DH * 2);
  void* qkvraw = alloc((size_t)MH * 768 * 2);  // aliased: qkv bf16 | heavy_out f32
  __bf16* qkvb = (__bf16*)qkvraw;
  float* heavyout = (float*)qkvraw;
  __bf16* attnout = (__bf16*)alloc((size_t)MH * DH * 2);
  __bf16* ffbuf = (__bf16*)alloc((size_t)MH * 1024 * 2);
  __bf16* pLin1 = (__bf16*)alloc((size_t)384 * 384 * 2);
  __bf16* pLin2 = (__bf16*)alloc((size_t)384 * 384 * 2);
  __bf16 *pInP[2], *pOutP[2], *pFF1[2], *pFF2[2];
  for (int l = 0; l < 2; l++) {
    pInP[l] = (__bf16*)alloc((size_t)256 * 768 * 2);
    pOutP[l] = (__bf16*)alloc((size_t)256 * 256 * 2);
    pFF1[l] = (__bf16*)alloc((size_t)256 * 1024 * 2);
    pFF2[l] = (__bf16*)alloc((size_t)1024 * 256 * 2);
  }
  __bf16* pPin = (__bf16*)alloc((size_t)384 * 256 * 2);
  __bf16* pPout = (__bf16*)alloc((size_t)256 * 384 * 2);

  auto packw = [&](const float* w, __bf16* dst, int OUT, int IN) {
    int n = OUT * IN;
    k_pack_linw<<<(n + 255) / 256, 256, 0, stream>>>(w, dst, OUT, IN);
  };
  auto gemm = [&](const __bf16* A, const __bf16* Bp, const float* bias, int M,
                  int Kd, int Nt, int mode, float* oF, __bf16* oB,
                  const float* res) {
    dim3 g(M / 128, Nt / 64);
    k_gemm<<<g, 256, 0, stream>>>(A, Bp, bias, oF, oB, res, M, Kd, Nt, mode);
  };

  // --------- weight packing + input cast ---------
  k_cast_bf16<<<(int)((BND + 255) / 256), 256, 0, stream>>>(feat_cur, actA, (int)BND);
  for (int l = 0; l < 3; l++)
    k_pack_convw<<<(3456 * 384 + 255) / 256, 256, 0, stream>>>(fwm_cw[l], convW[l]);
  packw(lin1_w, pLin1, 384, 384);
  packw(lin2_w, pLin2, 384, 384);
  for (int l = 0; l < 2; l++) {
    packw(inp_w[l], pInP[l], 768, 256);
    packw(op_w[l], pOutP[l], 256, 256);
    packw(ff1_w[l], pFF1[l], 1024, 256);
    packw(ff2_w[l], pFF2[l], 256, 1024);
  }
  packw(pin_w, pPin, 256, 384);
  packw(pout_w, pPout, 384, 256);

  // --------- FWM conv stack (WMMA) + fused surprise partials ---------
  dim3 cg(8, BATCH);
  k_conv_gn<<<cg, 256, 0, stream>>>(actA, convW[0], fwm_cb[0], fwm_gg[0], fwm_gb[0],
                                    actB, nullptr, nullptr, 1);
  k_conv_gn<<<cg, 256, 0, stream>>>(actB, convW[1], fwm_cb[1], fwm_gg[1], fwm_gb[1],
                                    actA, nullptr, nullptr, 1);
  k_conv_gn<<<cg, 256, 0, stream>>>(actA, convW[2], fwm_cb[2], fwm_gg[2], fwm_gb[2],
                                    nullptr, psq, feat_next, 0);

  // --------- surprise, loss, top-K ---------
  k_surprise<<<256, 256, 0, stream>>>(psq, surp, lpart);
  k_loss<<<1, 256, 0, stream>>>(lpart, loss);
  k_topk<<<BATCH, 256, 0, stream>>>(surp, topk, mask);

  // --------- light path: LN -> lin1+SiLU -> lin2 + residual ---------
  k_layernorm<<<BN / 8, 256, 0, stream>>>(feat_cur, lln_g, lln_b, actB, BN, DIM);
  gemm(actB, pLin1, lin1_b, BN, DIM, DIM, EP_SILU, nullptr, actA, nullptr);
  gemm(actA, pLin2, lin2_b, BN, DIM, DIM, EP_F32RES, routed, nullptr, feat_cur);

  // --------- heavy path on top-K tokens ---------
  k_gather<<<MH, 128, 0, stream>>>(feat_cur, topk, tok);
  gemm(tok, pPin, pin_b, MH, DIM, DH, EP_F32, hbuf, nullptr, nullptr);
  for (int l = 0; l < 2; l++) {
    k_layernorm<<<MH / 8, 256, 0, stream>>>(hbuf, l1g[l], l1b[l], lnbuf, MH, DH);
    gemm(lnbuf, pInP[l], inp_b[l], MH, DH, 768, EP_BF16, nullptr, qkvb, nullptr);
    k_attn<<<dim3(4, BATCH), 256, 0, stream>>>(qkvb, attnout);
    gemm(attnout, pOutP[l], op_b[l], MH, DH, DH, EP_F32RES, hbuf, nullptr, hbuf);
    k_layernorm<<<MH / 8, 256, 0, stream>>>(hbuf, l2g[l], l2b[l], lnbuf, MH, DH);
    gemm(lnbuf, pFF1[l], ff1_b[l], MH, DH, 1024, EP_GELU, nullptr, ffbuf, nullptr);
    gemm(ffbuf, pFF2[l], ff2_b[l], MH, 1024, DH, EP_F32RES, hbuf, nullptr, hbuf);
  }
  k_cast_bf16<<<(MH * DH + 255) / 256, 256, 0, stream>>>(hbuf, lnbuf, MH * DH);
  gemm(lnbuf, pPout, pout_b, MH, DH, DIM, EP_F32, heavyout, nullptr, nullptr);
  k_scatter<<<dim3(BATCH, KTOP), 128, 0, stream>>>(heavyout, topk, routed);
}
